// CausalMultiHeadSelfAttention_23390391894854
// MI455X (gfx1250) — matrix operand order repaired and checked by reading, not compiled, not run on hardware
//
#include <hip/hip_runtime.h>
#include <hip/hip_fp16.h>

// ---------------------------------------------------------------- constants
#define B_   2
#define S_   2048
#define DM   1024          // d_model
#define H_   16
#define DH   64            // d_head
#define BS   (B_ * S_)     // 4096 rows
#define N_QKV (3 * DM)     // 3072
#define PADS 40            // LDS tile row stride: 32 f16 + 8 pad (TDM pad feature)

typedef __attribute__((ext_vector_type(16))) _Float16 v16h;
typedef __attribute__((ext_vector_type(8)))  _Float16 v8h;
typedef __attribute__((ext_vector_type(4)))  _Float16 v4h;
typedef __attribute__((ext_vector_type(8)))  float    v8f;
typedef __attribute__((ext_vector_type(4)))  unsigned int v4u;
typedef __attribute__((ext_vector_type(8)))  int      v8i;
typedef __attribute__((ext_vector_type(4)))  int      v4i;

#if __has_builtin(__builtin_amdgcn_tensor_load_to_lds)
#define HAVE_TDM 1
#else
#define HAVE_TDM 0
#endif

#if __has_builtin(__builtin_amdgcn_s_wait_tensorcnt)
#define TENSOR_WAIT(n) __builtin_amdgcn_s_wait_tensorcnt(n)
#else
#define TENSOR_WAIT(n) asm volatile("s_wait_tensorcnt %0" ::"i"(n) : "memory")
#endif

// ------------------------------------------------------- fragment loaders
// WMMA 16-bit A-matrix 16x32 fragment in this code's contiguous k order (not ISA table order):
//   lane m (<16):  row m, elems[0..15]=K{0..15}
//   lane m+16:     row m, elems[0..15]=K{16..31}
__device__ __forceinline__ v16h frag_a_load(const _Float16* tile, int stride, int lane) {
    const _Float16* p = tile + (lane & 15) * stride + ((lane >> 4) << 4);
    v8h lo = *(const v8h*)(p);
    v8h hi = *(const v8h*)(p + 8);
    v16h f;
#pragma unroll
    for (int i = 0; i < 8; ++i) { f[i] = lo[i]; f[i + 8] = hi[i]; }
    return f;
}

// 16-bit B-matrix 32x16: lane n(<16): column n, K=0..15 contiguous;
// lane n+16: column n, K=16..31. (column n of B == row n of row-major [N,K])
__device__ __forceinline__ v16h frag_b_load(const _Float16* tile, int stride, int lane) {
    return *(const v16h*)(tile + (lane & 15) * stride + ((lane >> 4) << 4));
}
// same pattern, 16B-granular accesses (LDS-safe alignment with padded stride)
__device__ __forceinline__ v16h frag_b_load_lds(const _Float16* tile, int stride, int lane) {
    const _Float16* p = tile + (lane & 15) * stride + ((lane >> 4) << 4);
    v8h lo = *(const v8h*)(p);
    v8h hi = *(const v8h*)(p + 8);
    v16h f;
#pragma unroll
    for (int i = 0; i < 8; ++i) { f[i] = lo[i]; f[i + 8] = hi[i]; }
    return f;
}

__device__ __forceinline__ v8f wmma_f16(v16h a, v16h b, v8f c) {
    return __builtin_amdgcn_wmma_f32_16x16x32_f16(false, a, false, b, (short)0, c,
                                                  false, false);
}

__device__ __forceinline__ float half_max(float x) {
#pragma unroll
    for (int m = 8; m >= 1; m >>= 1) x = fmaxf(x, __shfl_xor(x, m, 32));
    return x;
}
__device__ __forceinline__ float half_sum(float x) {
#pragma unroll
    for (int m = 8; m >= 1; m >>= 1) x += __shfl_xor(x, m, 32);
    return x;
}

#if HAVE_TDM
// ---------------------------------------------------- TDM 2D tile descriptor
// D# per CDNA5 ISA ch.8: group0 = {count, lds_addr, global_addr, type=2},
// group1 = {data_size=2B, pad codes, tensor dims, tile dims, dim0 stride}.
// Pad: every 16 DWORDs (one 32-f16 row) insert 4 DWORDs -> LDS row stride 40.
// This toolchain's builtin is the 6-arg form:
//   (uint32x4 g0, int32x8 g1, int32x4 g2, int32x4 g3, int32x8 xtra, i32 cpol)
__device__ __forceinline__ void tdm_load_2d(unsigned lds_off, const void* gptr,
                                            unsigned tile_d0, unsigned tile_d1,
                                            unsigned tensor_d0, unsigned tensor_d1,
                                            unsigned stride_d0) {
    unsigned long long ga = (unsigned long long)gptr;
    v4u g0;
    g0[0] = 1u;                                   // count=1, user mode
    g0[1] = lds_off;                              // LDS byte address
    g0[2] = (unsigned)(ga & 0xffffffffu);         // global addr [31:0]
    g0[3] = (unsigned)((ga >> 32) & 0x01ffffffu)  // global addr [56:32]
            | (2u << 30);                         // type = 2 ("image")
    v8i g1;
    g1[0] = (int)((1u << 16)        // data_size = 1 -> 2 bytes
                  | (1u << 20)      // pad_enable
                  | (3u << 22)      // pad_interval code 3 -> every 16 DWORDs
                  | (3u << 25));    // pad_amount  code 3 -> 4 DWORDs (8 f16)
    g1[1] = (int)((tensor_d0 & 0xffffu) << 16);   // tensor_dim0[15:0]
    g1[2] = (int)(((tensor_d0 >> 16) & 0xffffu) | ((tensor_d1 & 0xffffu) << 16));
    g1[3] = (int)(((tensor_d1 >> 16) & 0xffffu) | ((tile_d0 & 0xffffu) << 16));
    g1[4] = (int)(tile_d1 & 0xffffu);             // tile_dim1 (tile_dim2 = 0)
    g1[5] = (int)stride_d0;                       // tensor_dim0_stride[31:0]
    g1[6] = 0;                                    // stride[47:32], dim1_stride lo
    g1[7] = 0;
    v4i gz4 = {0, 0, 0, 0};                       // 2D: groups 2/3 unused
    v8i gz8 = {0, 0, 0, 0, 0, 0, 0, 0};
    __builtin_amdgcn_tensor_load_to_lds(g0, g1, gz4, gz4, gz8, 0);
}
#endif

// ---------------------------------------------------------------- kernels
__global__ __launch_bounds__(256) void cvt_f32_f16_kernel(
        const float* __restrict__ src, _Float16* __restrict__ dst) {
    int i = (blockIdx.x * 256 + threadIdx.x) * 4;
    float4 v = *(const float4*)(src + i);
    v4h o;
    o[0] = (_Float16)v.x; o[1] = (_Float16)v.y;
    o[2] = (_Float16)v.z; o[3] = (_Float16)v.w;
    *(v4h*)(dst + i) = o;
}

#if !HAVE_TDM
// cooperative synchronous staging fallback (same padded LDS layout)
__device__ __forceinline__ void stage_tiles(_Float16* dA, _Float16* dB,
                                            const _Float16* A, const _Float16* W,
                                            int K, int tid) {
    {   // A: 128 rows x 32 f16, two threads per row
        int r = tid >> 1, h = tid & 1;
        const _Float16* s = A + (size_t)r * K + 16 * h;
        v8h x0 = *(const v8h*)(s);
        v8h x1 = *(const v8h*)(s + 8);
        _Float16* d = dA + r * PADS + 16 * h;
        *(v8h*)(d) = x0; *(v8h*)(d + 8) = x1;
    }
    if (tid < 128) {  // B: 64 rows x 32 f16
        int r = tid >> 1, h = tid & 1;
        const _Float16* s = W + (size_t)r * K + 16 * h;
        v8h x0 = *(const v8h*)(s);
        v8h x1 = *(const v8h*)(s + 8);
        _Float16* d = dB + r * PADS + 16 * h;
        *(v8h*)(d) = x0; *(v8h*)(d + 8) = x1;
    }
}
#endif

// C[M,N] = A[M,K] * W[N,K]^T.  Block = 8 waves = 128x64 tile; K staged in
// 32-wide LDS tiles, double-buffered via TDM DMA issued by wave 0.
// grid = (N/64, M/128)
template <bool OUT_F32>
__global__ __launch_bounds__(256) void gemm_nt_kernel(
        const _Float16* __restrict__ A, const _Float16* __restrict__ W,
        float* __restrict__ Cf, _Float16* __restrict__ Ch,
        int M, int N, int K) {
    __shared__ _Float16 lA[2][128 * PADS];
    __shared__ _Float16 lB[2][64 * PADS];
    const int lane = threadIdx.x & 31;
    const int wave = threadIdx.x >> 5;
    const int m0 = blockIdx.y * 128;
    const int n0 = blockIdx.x * 64;
    const int nk = K / 32;

#if HAVE_TDM
    const unsigned offA0 = (unsigned)(size_t)&lA[0][0];
    const unsigned offA1 = (unsigned)(size_t)&lA[1][0];
    const unsigned offB0 = (unsigned)(size_t)&lB[0][0];
    const unsigned offB1 = (unsigned)(size_t)&lB[1][0];
    if (wave == 0) {   // prologue: stage 0
        tdm_load_2d(offA0, A + (size_t)m0 * K, 32, 128, K, M, K);
        tdm_load_2d(offB0, W + (size_t)n0 * K, 32, 64, K, N, K);
    }
#endif

    v8f acc0 = {}, acc1 = {}, acc2 = {}, acc3 = {};
    for (int i = 0; i < nk; ++i) {
        const int cur = i & 1;
#if HAVE_TDM
        if (wave == 0) {
            if (i + 1 < nk) {   // prefetch next stage, then drain current
                tdm_load_2d(cur ? offA0 : offA1,
                            A + (size_t)m0 * K + (size_t)(i + 1) * 32, 32, 128, K, M, K);
                tdm_load_2d(cur ? offB0 : offB1,
                            W + (size_t)n0 * K + (size_t)(i + 1) * 32, 32, 64, K, N, K);
                TENSOR_WAIT(2);
            } else {
                TENSOR_WAIT(0);
            }
        }
#else
        stage_tiles(&lA[cur][0], &lB[cur][0],
                    A + (size_t)m0 * K + (size_t)i * 32,
                    W + (size_t)n0 * K + (size_t)i * 32, K, threadIdx.x);
#endif
        __syncthreads();
        const _Float16* ta = &lA[cur][0] + wave * 16 * PADS;
        const _Float16* tb = &lB[cur][0];
        v16h a  = frag_a_load(ta, PADS, lane);
        acc0 = wmma_f16(a, frag_b_load_lds(tb + 0 * 16 * PADS, PADS, lane), acc0);
        acc1 = wmma_f16(a, frag_b_load_lds(tb + 1 * 16 * PADS, PADS, lane), acc1);
        acc2 = wmma_f16(a, frag_b_load_lds(tb + 2 * 16 * PADS, PADS, lane), acc2);
        acc3 = wmma_f16(a, frag_b_load_lds(tb + 3 * 16 * PADS, PADS, lane), acc3);
        __syncthreads();   // protect buffer before it is re-filled
    }

    // C/D layout: row = m0+wave*16 + v + 8*(lane>=16), col = n0 + (lane&15)
    const int col = lane & 15;
    const int rb  = m0 + wave * 16 + ((lane >> 4) << 3);
#pragma unroll
    for (int v = 0; v < 8; ++v) {
        size_t r = (size_t)(rb + v) * N;
        if (OUT_F32) {
            Cf[r + n0 +  0 + col] = acc0[v];
            Cf[r + n0 + 16 + col] = acc1[v];
            Cf[r + n0 + 32 + col] = acc2[v];
            Cf[r + n0 + 48 + col] = acc3[v];
        } else {
            Ch[r + n0 +  0 + col] = (_Float16)acc0[v];
            Ch[r + n0 + 16 + col] = (_Float16)acc1[v];
            Ch[r + n0 + 32 + col] = (_Float16)acc2[v];
            Ch[r + n0 + 48 + col] = (_Float16)acc3[v];
        }
    }
}

// RoPE on Q,K + head split; V stored transposed [B,H,DH,S].
__global__ __launch_bounds__(256) void rope_split_kernel(
        const _Float16* __restrict__ qkv, const float* __restrict__ cosb,
        const float* __restrict__ sinb, const int* __restrict__ tpos,
        _Float16* __restrict__ Qh, _Float16* __restrict__ Kh,
        _Float16* __restrict__ Vt) {
    int idx = blockIdx.x * 256 + threadIdx.x;   // [0, B*S*H*32)
    int p = idx & 31;
    int h = (idx >> 5) & (H_ - 1);
    int s = (idx >> 9) & (S_ - 1);
    int b = idx >> 20;
    int pos = tpos[s];
    float c  = cosb[pos * 32 + p];
    float sn = sinb[pos * 32 + p];
    const _Float16* row = qkv + (size_t)(b * S_ + s) * N_QKV + h * DH;
    float qe = (float)row[2 * p],          qo = (float)row[2 * p + 1];
    float ke = (float)row[DM + 2 * p],     ko = (float)row[DM + 2 * p + 1];
    float ve = (float)row[2 * DM + 2 * p], vo = (float)row[2 * DM + 2 * p + 1];
    size_t hb = (size_t)(b * H_ + h) * (S_ * DH);
    size_t so = hb + (size_t)s * DH;
    Qh[so + 2 * p]     = (_Float16)(qe * c - qo * sn);
    Qh[so + 2 * p + 1] = (_Float16)(qe * sn + qo * c);
    Kh[so + 2 * p]     = (_Float16)(ke * c - ko * sn);
    Kh[so + 2 * p + 1] = (_Float16)(ke * sn + ko * c);
    Vt[hb + (size_t)(2 * p) * S_ + s]     = (_Float16)ve;
    Vt[hb + (size_t)(2 * p + 1) * S_ + s] = (_Float16)vo;
}

// Flash attention: grid = (S/128, B*H); 8 waves/block, wave owns 16 q rows.
__global__ __launch_bounds__(256) void flash_attn_kernel(
        const _Float16* __restrict__ Qh, const _Float16* __restrict__ Kh,
        const _Float16* __restrict__ Vt, _Float16* __restrict__ Y /*[BS,DM]*/) {
    __shared__ _Float16 plds[8][16 * 32];   // wave-private P staging tiles
    const int lane = threadIdx.x & 31;
    const int wave = threadIdx.x >> 5;
    const int bh = blockIdx.y;
    const int b = bh >> 4, h = bh & 15;
    const int q0 = blockIdx.x * 128 + wave * 16;

    const _Float16* Q  = Qh + (size_t)bh * (S_ * DH);
    const _Float16* Kp = Kh + (size_t)bh * (S_ * DH);
    const _Float16* Vp = Vt + (size_t)bh * (DH * S_);
    _Float16* my = &plds[wave][0];

    v16h qa0 = frag_a_load(Q + (size_t)q0 * DH, DH, lane);
    v16h qa1 = frag_a_load(Q + (size_t)q0 * DH + 32, DH, lane);

    v8f o0 = {}, o1 = {}, o2 = {}, o3 = {};
    float mi[8], li[8];
#pragma unroll
    for (int v = 0; v < 8; ++v) { mi[v] = -1e30f; li[v] = 0.0f; }

    const float sc = 0.125f * 1.44269504088896f;   // 1/sqrt(64) * log2(e)
    const int colb  = lane & 15;
    const int rbase = q0 + ((lane >> 4) << 3);
    const int kv_end = (q0 + 16 + 31) & ~31;       // causal: own diagonal only

    for (int j = 0; j < kv_end; j += 32) {
        v8f s0 = {}, s1 = {};
        s0 = wmma_f16(qa0, frag_b_load(Kp + (size_t)j * DH,             DH, lane), s0);
        s0 = wmma_f16(qa1, frag_b_load(Kp + (size_t)j * DH + 32,        DH, lane), s0);
        s1 = wmma_f16(qa0, frag_b_load(Kp + (size_t)(j + 16) * DH,      DH, lane), s1);
        s1 = wmma_f16(qa1, frag_b_load(Kp + (size_t)(j + 16) * DH + 32, DH, lane), s1);

        const int c0 = j + colb, c1 = j + 16 + colb;
#pragma unroll
        for (int v = 0; v < 8; ++v) {
            const int r = rbase + v;
            float a = s0[v] * sc, bb = s1[v] * sc;
            if (c0 > r) a  = -3.0e38f;
            if (c1 > r) bb = -3.0e38f;
            s0[v] = a; s1[v] = bb;
        }
#pragma unroll
        for (int v = 0; v < 8; ++v) {
            float rm = half_max(fmaxf(s0[v], s1[v]));
            float mn = fmaxf(mi[v], rm);
            float alpha = exp2f(mi[v] - mn);
            float p0 = exp2f(s0[v] - mn);
            float p1 = exp2f(s1[v] - mn);
            float rs = half_sum(p0 + p1);
            li[v] = li[v] * alpha + rs;
            mi[v] = mn;
            o0[v] *= alpha; o1[v] *= alpha; o2[v] *= alpha; o3[v] *= alpha;
            const int r = ((lane >> 4) << 3) + v;
            my[r * 32 + colb]      = (_Float16)p0;
            my[r * 32 + 16 + colb] = (_Float16)p1;
        }
        asm volatile("s_wait_dscnt 0" ::: "memory");   // C->A layout via LDS
        v16h pa = frag_a_load(my, 32, lane);
        o0 = wmma_f16(pa, frag_b_load(Vp + (size_t) 0 * S_ + j, S_, lane), o0);
        o1 = wmma_f16(pa, frag_b_load(Vp + (size_t)16 * S_ + j, S_, lane), o1);
        o2 = wmma_f16(pa, frag_b_load(Vp + (size_t)32 * S_ + j, S_, lane), o2);
        o3 = wmma_f16(pa, frag_b_load(Vp + (size_t)48 * S_ + j, S_, lane), o3);
    }

#pragma unroll
    for (int v = 0; v < 8; ++v) {
        const float inv = 1.0f / li[v];
        const int r = rbase + v;
        size_t base = (size_t)(b * S_ + r) * DM + h * DH;
        Y[base +  0 + colb] = (_Float16)(o0[v] * inv);
        Y[base + 16 + colb] = (_Float16)(o1[v] * inv);
        Y[base + 32 + colb] = (_Float16)(o2[v] * inv);
        Y[base + 48 + colb] = (_Float16)(o3[v] * inv);
    }
}

// ---------------------------------------------------------------- launcher
extern "C" void kernel_launch(void* const* d_in, const int* in_sizes, int n_in,
                              void* d_out, int out_size, void* d_ws, size_t ws_size,
                              hipStream_t stream) {
    (void)in_sizes; (void)n_in; (void)out_size; (void)ws_size;
    const float* x    = (const float*)d_in[0];
    const float* wqkv = (const float*)d_in[1];
    const float* wout = (const float*)d_in[2];
    const float* cosb = (const float*)d_in[3];
    const float* sinb = (const float*)d_in[4];
    const int*   tpos = (const int*)d_in[5];
    float* out = (float*)d_out;

    char* ws = (char*)d_ws;
    size_t off = 0;
    auto alloc = [&](size_t bytes) {
        void* p = ws + off;
        off += (bytes + 255) & ~(size_t)255;
        return p;
    };
    _Float16* x16   = (_Float16*)alloc((size_t)BS * DM * 2);
    _Float16* w1_16 = (_Float16*)alloc((size_t)N_QKV * DM * 2);
    _Float16* w2_16 = (_Float16*)alloc((size_t)DM * DM * 2);
    _Float16* qkv16 = (_Float16*)alloc((size_t)BS * N_QKV * 2);
    _Float16* Qhd   = (_Float16*)alloc((size_t)B_ * H_ * S_ * DH * 2);
    _Float16* Khd   = (_Float16*)alloc((size_t)B_ * H_ * S_ * DH * 2);
    _Float16* Vtd   = (_Float16*)alloc((size_t)B_ * H_ * DH * S_ * 2);
    _Float16* Y16   = (_Float16*)alloc((size_t)BS * DM * 2);

    cvt_f32_f16_kernel<<<(BS * DM) / 1024, 256, 0, stream>>>(x, x16);
    cvt_f32_f16_kernel<<<(N_QKV * DM) / 1024, 256, 0, stream>>>(wqkv, w1_16);
    cvt_f32_f16_kernel<<<(DM * DM) / 1024, 256, 0, stream>>>(wout, w2_16);

    gemm_nt_kernel<false><<<dim3(N_QKV / 64, BS / 128), 256, 0, stream>>>(
        x16, w1_16, nullptr, qkv16, BS, N_QKV, DM);

    rope_split_kernel<<<(B_ * S_ * H_ * 32) / 256, 256, 0, stream>>>(
        qkv16, cosb, sinb, tpos, Qhd, Khd, Vtd);

    flash_attn_kernel<<<dim3(S_ / 128, B_ * H_), 256, 0, stream>>>(
        Qhd, Khd, Vtd, Y16);

    gemm_nt_kernel<true><<<dim3(DM / 64, BS / 128), 256, 0, stream>>>(
        Y16, w2_16, out, nullptr, BS, DM, DM);
}
